// TextTaskPromptCrossAttention_16312285790680
// MI455X (gfx1250) — matrix operand order repaired and checked by reading, not compiled, not run on hardware
//
#include <hip/hip_runtime.h>

// ---------------------------------------------------------------------------
// CDNA5 (gfx1250) wave32 WMMA implementation of TextTaskPromptCrossAttention.
// All matmuls: v_wmma_f32_16x16x32_bf16 (bf16 inputs, fp32 accumulate).
// Intermediates (q, img/prompt ctx, kv-ctx, k, v) are carried in global memory
// as bf16: GEMM epilogues convert for free, and consumers stage tiles into LDS
// as pure b128 copies (bf16 pairs along the reduction dim are contiguous).
// ---------------------------------------------------------------------------

typedef __attribute__((ext_vector_type(16))) __bf16   v16bf;
typedef __attribute__((ext_vector_type(8)))  float    v8f;
typedef __attribute__((ext_vector_type(8)))  unsigned v8u;
typedef __attribute__((ext_vector_type(4)))  unsigned u32x4;
typedef __attribute__((ext_vector_type(4)))  float    f32x4;

#define TCD     768
#define CTXL    77
#define NPROMPT 64
#define NIMG    2048
#define JREAL   2189      // 77 + 64 + 2048
#define JPAD    2208      // multiple of 32; 2*JPAD = 4416 = 69*64
#define INNER   512
#define DHEAD   64

// pack two fp32 -> bf16x2 u32 (round-to-nearest): 2x v_add + 1x v_perm_b32
__device__ __forceinline__ unsigned bf16pk(float lo, float hi) {
  unsigned a = __builtin_bit_cast(unsigned, lo) + 0x8000u;
  unsigned b = __builtin_bit_cast(unsigned, hi) + 0x8000u;
#if __has_builtin(__builtin_amdgcn_perm)
  return __builtin_amdgcn_perm(b, a, 0x07060302u);   // {hi16(b), hi16(a)}
#else
  return (a >> 16) | (b & 0xFFFF0000u);
#endif
}
__device__ __forceinline__ unsigned perm_lo16(unsigned b, unsigned a) { // {b.lo16, a.lo16}
#if __has_builtin(__builtin_amdgcn_perm)
  return __builtin_amdgcn_perm(b, a, 0x05040100u);
#else
  return (a & 0xFFFFu) | (b << 16);
#endif
}
__device__ __forceinline__ unsigned perm_hi16(unsigned b, unsigned a) { // {b.hi16, a.hi16}
#if __has_builtin(__builtin_amdgcn_perm)
  return __builtin_amdgcn_perm(b, a, 0x07060302u);
#else
  return (a >> 16) | (b & 0xFFFF0000u);
#endif
}

__device__ __forceinline__ v16bf mkfrag(u32x4 a, u32x4 b) {
  v8u u;
  u[0] = a[0]; u[1] = a[1]; u[2] = a[2]; u[3] = a[3];
  u[4] = b[0]; u[5] = b[1]; u[6] = b[2]; u[7] = b[3];
  return __builtin_bit_cast(v16bf, u);
}

// ---------------------------------------------------------------------------
// GEMM core pieces. Block tile 64x64, K step 32, 256 threads = 8 waves.
// Requires M%64==0, N%64==0, K%32==0.
// sA: [row][pair] packed bf16x2 along K  -> A-frag = 2x ds_load_b128
// sB: [col][pair] transposed-packed      -> B-frag = 2x ds_load_b128
// ---------------------------------------------------------------------------
#define GBK  32
#define APAD 20
#define BPAD 20

template <bool CBF>
__device__ __forceinline__ void gemm_epilogue(void* __restrict__ Cv,
    const float* __restrict__ bias, int N, int rb, int nc0,
    const v8f& acc0, const v8f& acc1)
{
  if constexpr (CBF) {
    unsigned short* Cb = (unsigned short*)Cv;
    #pragma unroll
    for (int r = 0; r < 8; ++r) {
      unsigned u0 = (__builtin_bit_cast(unsigned, acc0[r]) + 0x8000u) >> 16;
      unsigned u1 = (__builtin_bit_cast(unsigned, acc1[r]) + 0x8000u) >> 16;
      Cb[(size_t)(rb + r) * N + nc0]      = (unsigned short)u0;
      Cb[(size_t)(rb + r) * N + nc0 + 16] = (unsigned short)u1;
    }
  } else {
    float* Cf = (float*)Cv;
    const float b0 = bias ? bias[nc0]      : 0.0f;
    const float b1 = bias ? bias[nc0 + 16] : 0.0f;
    #pragma unroll
    for (int r = 0; r < 8; ++r) {
      Cf[(size_t)(rb + r) * N + nc0]      = acc0[r] + b0;
      Cf[(size_t)(rb + r) * N + nc0 + 16] = acc1[r] + b1;
    }
  }
}

// -------- GEMM with fp32 A --------
template <bool CBF>
__global__ __launch_bounds__(256) void gemm_af32(
    const float* __restrict__ A, const float* __restrict__ B,
    void* __restrict__ Cv, const float* __restrict__ bias,
    int M, int N, int K)
{
  __shared__ unsigned sA[64 * APAD];
  __shared__ unsigned sB[64 * BPAD];
  const int tid  = threadIdx.x;
  const int lane = tid & 31;
  const int wave = tid >> 5;
  const int wm = wave & 3, wn = wave >> 2;
  const int l15 = lane & 15;
  const bool hi = lane >= 16;
  const int blockM = blockIdx.y * 64;
  const int blockN = blockIdx.x * 64;

  const float* aP0 = A + (size_t)(blockM + (tid >> 3)) * K + 4 * (tid & 7);
  const float* aP1 = aP0 + (size_t)32 * K;
  const float* bP0 = B + (size_t)(2 * (tid >> 6)) * N + blockN + (tid & 63);
  const float* bP1 = bP0 + (size_t)8 * N;
  const float* bP2 = bP1 + (size_t)8 * N;
  const float* bP3 = bP2 + (size_t)8 * N;
  const size_t bStep = (size_t)GBK * N;
  const int sAi = (tid >> 3) * APAD + 2 * (tid & 7);
  const int sBi = (tid & 63) * BPAD + (tid >> 6);

  v8f acc0 = {}; v8f acc1 = {};

  for (int kk = 0; kk < K; kk += GBK) {
    f32x4 va0 = *(const f32x4*)aP0;
    f32x4 va1 = *(const f32x4*)aP1;
    float b00 = bP0[0], b01 = bP0[N];
    float b10 = bP1[0], b11 = bP1[N];
    float b20 = bP2[0], b21 = bP2[N];
    float b30 = bP3[0], b31 = bP3[N];

    sA[sAi]               = bf16pk(va0[0], va0[1]);
    sA[sAi + 1]           = bf16pk(va0[2], va0[3]);
    sA[sAi + 32*APAD]     = bf16pk(va1[0], va1[1]);
    sA[sAi + 32*APAD + 1] = bf16pk(va1[2], va1[3]);
    sB[sBi]      = bf16pk(b00, b01);
    sB[sBi + 4]  = bf16pk(b10, b11);
    sB[sBi + 8]  = bf16pk(b20, b21);
    sB[sBi + 12] = bf16pk(b30, b31);
    __syncthreads();

    if (kk + GBK < K) __builtin_prefetch(aP0 + GBK, 0, 1);
    aP0 += GBK; aP1 += GBK;
    bP0 += bStep; bP1 += bStep; bP2 += bStep; bP3 += bStep;

    const unsigned* ar = &sA[(wm * 16 + l15) * APAD + (hi ? 4 : 0)];
    v16bf af = mkfrag(*(const u32x4*)ar, *(const u32x4*)(ar + 8));
    const unsigned* br0 = &sB[(wn * 32 + l15) * BPAD + (hi ? 4 : 0)];
    v16bf bf0 = mkfrag(*(const u32x4*)br0, *(const u32x4*)(br0 + 8));
    const unsigned* br1 = br0 + 16 * BPAD;
    v16bf bf1 = mkfrag(*(const u32x4*)br1, *(const u32x4*)(br1 + 8));

    acc0 = __builtin_amdgcn_wmma_f32_16x16x32_bf16(false, af, false, bf0, (short)0, acc0, false, false);
    acc1 = __builtin_amdgcn_wmma_f32_16x16x32_bf16(false, af, false, bf1, (short)0, acc1, false, false);
    __syncthreads();
  }
  gemm_epilogue<CBF>(Cv, bias, N, blockM + wm * 16 + (hi ? 8 : 0),
                     blockN + wn * 32 + l15, acc0, acc1);
}

// -------- GEMM with bf16 A (staging = pure b128 copies) --------
template <bool CBF>
__global__ __launch_bounds__(256) void gemm_abf16(
    const __bf16* __restrict__ A, const float* __restrict__ B,
    void* __restrict__ Cv, const float* __restrict__ bias,
    int M, int N, int K)
{
  __shared__ unsigned sA[64 * APAD];
  __shared__ unsigned sB[64 * BPAD];
  const int tid  = threadIdx.x;
  const int lane = tid & 31;
  const int wave = tid >> 5;
  const int wm = wave & 3, wn = wave >> 2;
  const int l15 = lane & 15;
  const bool hi = lane >= 16;
  const int blockM = blockIdx.y * 64;
  const int blockN = blockIdx.x * 64;

  // one 16B chunk (8 bf16 = 4 pairs) per thread per K-step: 64 rows x 4 chunks
  const __bf16* aP = A + (size_t)(blockM + (tid >> 2)) * K + 8 * (tid & 3);
  const float* bP0 = B + (size_t)(2 * (tid >> 6)) * N + blockN + (tid & 63);
  const float* bP1 = bP0 + (size_t)8 * N;
  const float* bP2 = bP1 + (size_t)8 * N;
  const float* bP3 = bP2 + (size_t)8 * N;
  const size_t bStep = (size_t)GBK * N;
  const int sAi = (tid >> 2) * APAD + 4 * (tid & 3);
  const int sBi = (tid & 63) * BPAD + (tid >> 6);

  v8f acc0 = {}; v8f acc1 = {};

  for (int kk = 0; kk < K; kk += GBK) {
    u32x4 aq = *(const u32x4*)aP;
    float b00 = bP0[0], b01 = bP0[N];
    float b10 = bP1[0], b11 = bP1[N];
    float b20 = bP2[0], b21 = bP2[N];
    float b30 = bP3[0], b31 = bP3[N];

    *(u32x4*)&sA[sAi] = aq;                 // ds_store_b128, no conversion
    sB[sBi]      = bf16pk(b00, b01);
    sB[sBi + 4]  = bf16pk(b10, b11);
    sB[sBi + 8]  = bf16pk(b20, b21);
    sB[sBi + 12] = bf16pk(b30, b31);
    __syncthreads();

    if (kk + GBK < K) __builtin_prefetch(aP + GBK, 0, 1);
    aP += GBK;
    bP0 += bStep; bP1 += bStep; bP2 += bStep; bP3 += bStep;

    const unsigned* ar = &sA[(wm * 16 + l15) * APAD + (hi ? 4 : 0)];
    v16bf af = mkfrag(*(const u32x4*)ar, *(const u32x4*)(ar + 8));
    const unsigned* br0 = &sB[(wn * 32 + l15) * BPAD + (hi ? 4 : 0)];
    v16bf bf0 = mkfrag(*(const u32x4*)br0, *(const u32x4*)(br0 + 8));
    const unsigned* br1 = br0 + 16 * BPAD;
    v16bf bf1 = mkfrag(*(const u32x4*)br1, *(const u32x4*)(br1 + 8));

    acc0 = __builtin_amdgcn_wmma_f32_16x16x32_bf16(false, af, false, bf0, (short)0, acc0, false, false);
    acc1 = __builtin_amdgcn_wmma_f32_16x16x32_bf16(false, af, false, bf1, (short)0, acc1, false, false);
    __syncthreads();
  }
  gemm_epilogue<CBF>(Cv, bias, N, blockM + wm * 16 + (hi ? 8 : 0),
                     blockN + wn * 32 + l15, acc0, acc1);
}

// ---------------------------------------------------------------------------
// Padded bf16 kv-context: [2][JPAD][TCD] = [cvt(text); prompt_bf; img_bf; 0]
// 16B (8 bf16) chunks.
// ---------------------------------------------------------------------------
__global__ void build_ctx_bf16(const float* __restrict__ text,
                               const __bf16* __restrict__ pctx,
                               const __bf16* __restrict__ img,
                               __bf16* __restrict__ out)
{
  size_t idx = (size_t)blockIdx.x * blockDim.x + threadIdx.x;   // chunk index
  const size_t total = (size_t)2 * JPAD * (TCD / 8);
  if (idx >= total) return;
  int    c8  = (int)(idx % (TCD / 8));
  size_t row = idx / (TCD / 8);
  int    j   = (int)(row % JPAD);
  int    b   = (int)(row / JPAD);
  u32x4 v = {0u, 0u, 0u, 0u};
  if (j < CTXL) {
    const float* s = &text[((size_t)b * CTXL + j) * TCD + 8 * c8];
    f32x4 x0 = *(const f32x4*)s;
    f32x4 x1 = *(const f32x4*)(s + 4);
    v[0] = bf16pk(x0[0], x0[1]); v[1] = bf16pk(x0[2], x0[3]);
    v[2] = bf16pk(x1[0], x1[1]); v[3] = bf16pk(x1[2], x1[3]);
  } else if (j < CTXL + NPROMPT) {
    v = *(const u32x4*)&pctx[(size_t)(j - CTXL) * TCD + 8 * c8];
  } else if (j < JREAL) {
    v = *(const u32x4*)&img[((size_t)b * NIMG + (j - CTXL - NPROMPT)) * TCD + 8 * c8];
  }
  *(u32x4*)&out[row * TCD + 8 * c8] = v;
}

// ---------------------------------------------------------------------------
// Flash attention (transposed-S). Q/K/V are bf16; staging is copy-only except
// V (2x v_perm per word). 128 threads = 4 waves.
// ---------------------------------------------------------------------------
#define KPAD 36
#define VPAD 20

__global__ __launch_bounds__(128) void attention_wmma(
    const __bf16* __restrict__ Q, const __bf16* __restrict__ Km,
    const __bf16* __restrict__ Vm, float* __restrict__ Om)
{
  __shared__ unsigned sK[32 * KPAD];  // [j_local][d-pair] = raw bf16 K rows
  __shared__ unsigned sV[64 * VPAD];  // [d][j-pair] transposed-packed
  const int tid  = threadIdx.x;
  const int lane = tid & 31;
  const int wave = tid >> 5;
  const int l15  = lane & 15;
  const bool hi  = lane >= 16;
  const int h       = blockIdx.y;
  const int rowBase = blockIdx.x * 64 + wave * 16;
  const int b       = rowBase >> 11;
  const float scale = 0.125f;
  const size_t kvbase = (size_t)b * JPAD * INNER + h * DHEAD;

  // staging pointers; in-loop offsets are immediates
  const __bf16* kP = Km + kvbase + (size_t)(tid >> 3) * INNER + 8 * (tid & 7);
  const __bf16* vP = Vm + kvbase + (size_t)(2 * (tid >> 5)) * INNER + 2 * (tid & 31);
  const int sKi = (tid >> 3) * KPAD + 4 * (tid & 7);
  const int sVi0 = (2 * (tid & 31)) * VPAD + (tid >> 5);

  // Q as B-operand: 16 consecutive bf16 per lane per half = 2x b128, raw.
  v16bf qf[2];
  {
    const __bf16* qr = Q + (size_t)(rowBase + l15) * INNER + h * DHEAD + (hi ? 16 : 0);
    qf[0] = mkfrag(*(const u32x4*)qr,        *(const u32x4*)(qr + 8));
    qf[1] = mkfrag(*(const u32x4*)(qr + 32), *(const u32x4*)(qr + 40));
  }

  float m = -1e30f, lsum = 0.0f;
  v8f o0 = {}, o1 = {}, o2 = {}, o3 = {};

  for (int jb = 0; jb < JPAD; jb += 32) {
    __syncthreads();
    // K tile: 32 rows x 64 bf16 = 256 chunks of 16B -> 2 copies per thread
    u32x4 k0 = *(const u32x4*)kP;
    u32x4 k1 = *(const u32x4*)(kP + (size_t)16 * INNER);
    // V tile: interleave rows 2p,2p+1 -> 4 units per thread (2 words each)
    unsigned va[4], vb[4];
    #pragma unroll
    for (int i = 0; i < 4; ++i) {
      va[i] = *(const unsigned*)(vP + (size_t)(8 * i) * INNER);
      vb[i] = *(const unsigned*)(vP + (size_t)(8 * i) * INNER + INNER);
    }
    *(u32x4*)&sK[sKi]             = k0;
    *(u32x4*)&sK[sKi + 16 * KPAD] = k1;
    #pragma unroll
    for (int i = 0; i < 4; ++i) {
      sV[sVi0 + 4 * i]        = perm_lo16(vb[i], va[i]);
      sV[sVi0 + VPAD + 4 * i] = perm_hi16(vb[i], va[i]);
    }
    __syncthreads();

    kP += (size_t)32 * INNER;
    vP += (size_t)32 * INNER;

    // ---- S^T tiles ----
    v8f s0 = {}, s1 = {};
    #pragma unroll
    for (int hh = 0; hh < 2; ++hh) {
      const unsigned* k0p = &sK[l15 * KPAD + hh * 16 + (hi ? 8 : 0)];
      v16bf kf0 = mkfrag(*(const u32x4*)k0p, *(const u32x4*)(k0p + 4));
      s0 = __builtin_amdgcn_wmma_f32_16x16x32_bf16(false, kf0, false, qf[hh], (short)0, s0, false, false);
      const unsigned* k1p = k0p + 16 * KPAD;
      v16bf kf1 = mkfrag(*(const u32x4*)k1p, *(const u32x4*)(k1p + 4));
      s1 = __builtin_amdgcn_wmma_f32_16x16x32_bf16(false, kf1, false, qf[hh], (short)0, s1, false, false);
    }

    // ---- scale + mask padded j ----
    float sv0[8], sv1[8];
    const int jo = hi ? 8 : 0;
    #pragma unroll
    for (int r = 0; r < 8; ++r) {
      sv0[r] = (jb + jo + r      < JREAL) ? s0[r] * scale : -1e30f;
      sv1[r] = (jb + 16 + jo + r < JREAL) ? s1[r] * scale : -1e30f;
    }

    // ---- online softmax ----
    float mx = -1e30f;
    #pragma unroll
    for (int r = 0; r < 8; ++r) mx = fmaxf(mx, fmaxf(sv0[r], sv1[r]));
    mx = fmaxf(mx, __shfl_xor(mx, 16, 32));
    const float mnew  = fmaxf(m, mx);
    const float alpha = __expf(m - mnew);
    float p0[8], p1[8], rs = 0.0f;
    #pragma unroll
    for (int r = 0; r < 8; ++r) {
      p0[r] = __expf(sv0[r] - mnew);
      p1[r] = __expf(sv1[r] - mnew);
      rs += p0[r] + p1[r];
    }
    rs += __shfl_xor(rs, 16, 32);
    lsum = lsum * alpha + rs;
    m = mnew;

    #pragma unroll
    for (int r = 0; r < 8; ++r) {
      float ar = __shfl(alpha, hi ? (r + 8) : r, 32);
      o0[r] *= ar; o1[r] *= ar; o2[r] *= ar; o3[r] *= ar;
    }

    // ---- P (C-layout == A-operand layout) ----
    v8u up;
    #pragma unroll
    for (int i = 0; i < 4; ++i) {
      up[i]     = bf16pk(p0[2*i], p0[2*i + 1]);
      up[4 + i] = bf16pk(p1[2*i], p1[2*i + 1]);
    }
    v16bf pf = __builtin_bit_cast(v16bf, up);

    // ---- O += P @ V ----
    {
      const unsigned* vr0 = &sV[l15 * VPAD + (hi ? 4 : 0)];
      const unsigned* vr1 = vr0 + 16 * VPAD;
      const unsigned* vr2 = vr1 + 16 * VPAD;
      const unsigned* vr3 = vr2 + 16 * VPAD;
      v16bf vf0 = mkfrag(*(const u32x4*)vr0, *(const u32x4*)(vr0 + 8));
      v16bf vf1 = mkfrag(*(const u32x4*)vr1, *(const u32x4*)(vr1 + 8));
      v16bf vf2 = mkfrag(*(const u32x4*)vr2, *(const u32x4*)(vr2 + 8));
      v16bf vf3 = mkfrag(*(const u32x4*)vr3, *(const u32x4*)(vr3 + 8));
      o0 = __builtin_amdgcn_wmma_f32_16x16x32_bf16(false, pf, false, vf0, (short)0, o0, false, false);
      o1 = __builtin_amdgcn_wmma_f32_16x16x32_bf16(false, pf, false, vf1, (short)0, o1, false, false);
      o2 = __builtin_amdgcn_wmma_f32_16x16x32_bf16(false, pf, false, vf2, (short)0, o2, false, false);
      o3 = __builtin_amdgcn_wmma_f32_16x16x32_bf16(false, pf, false, vf3, (short)0, o3, false, false);
    }
  }

  // ---- normalize and store (fp32 for the final GEMM's A) ----
  #pragma unroll
  for (int r = 0; r < 8; ++r) {
    float lr  = __shfl(lsum, hi ? (r + 8) : r, 32);
    float inv = 1.0f / lr;
    int row = rowBase + (hi ? 8 : 0) + r;
    float* op = Om + (size_t)row * INNER + h * DHEAD + l15;
    op[0]  = o0[r] * inv;
    op[16] = o1[r] * inv;
    op[32] = o2[r] * inv;
    op[48] = o3[r] * inv;
  }
}

// ---------------------------------------------------------------------------
// launch
// ---------------------------------------------------------------------------
extern "C" void kernel_launch(void* const* d_in, const int* in_sizes, int n_in,
                              void* d_out, int out_size, void* d_ws, size_t ws_size,
                              hipStream_t stream) {
  (void)in_sizes; (void)n_in; (void)out_size; (void)ws_size;
  const float* x        = (const float*)d_in[0];
  const float* context  = (const float*)d_in[1];
  const float* prompt   = (const float*)d_in[2];
  const float* w_prompt = (const float*)d_in[3];
  const float* w_img    = (const float*)d_in[4];
  const float* w_q      = (const float*)d_in[5];
  const float* w_k      = (const float*)d_in[6];
  const float* w_v      = (const float*)d_in[7];
  const float* w_out    = (const float*)d_in[8];
  const float* b_out    = (const float*)d_in[9];
  float* out = (float*)d_out;

  char* w = (char*)d_ws;
  __bf16* q_bf    = (__bf16*)w;  w += (size_t)4096 * 512 * 2;
  __bf16* img_bf  = (__bf16*)w;  w += (size_t)4096 * 768 * 2;
  __bf16* pctx_bf = (__bf16*)w;  w += (size_t)64 * 768 * 2;
  __bf16* ctx_bf  = (__bf16*)w;  w += (size_t)2 * JPAD * 768 * 2;
  __bf16* k_bf    = (__bf16*)w;  w += (size_t)2 * JPAD * 512 * 2;
  __bf16* v_bf    = (__bf16*)w;  w += (size_t)2 * JPAD * 512 * 2;
  float*  attn    = (float*)w;

  dim3 blk(256);
  // q = x @ w_q -> bf16              [4096,1024] x [1024,512]
  gemm_af32<true><<<dim3(512/64, 4096/64), blk, 0, stream>>>(x, w_q, q_bf, nullptr, 4096, 512, 1024);
  // img_ctx = x @ w_img -> bf16      [4096,1024] x [1024,768]
  gemm_af32<true><<<dim3(768/64, 4096/64), blk, 0, stream>>>(x, w_img, img_bf, nullptr, 4096, 768, 1024);
  // prompt_ctx -> bf16               [64,1024] x [1024,768]
  gemm_af32<true><<<dim3(768/64, 1), blk, 0, stream>>>(prompt, w_prompt, pctx_bf, nullptr, 64, 768, 1024);
  // ctx (bf16) = concat + pad
  {
    size_t ctotal = (size_t)2 * JPAD * (TCD / 8);
    build_ctx_bf16<<<(unsigned)((ctotal + 255) / 256), blk, 0, stream>>>(context, pctx_bf, img_bf, ctx_bf);
  }
  // k, v = ctx(bf16) @ w_k/w_v -> bf16   [4416,768] x [768,512]
  gemm_abf16<true><<<dim3(512/64, (2*JPAD)/64), blk, 0, stream>>>(ctx_bf, w_k, k_bf, nullptr, 2*JPAD, 512, 768);
  gemm_abf16<true><<<dim3(512/64, (2*JPAD)/64), blk, 0, stream>>>(ctx_bf, w_v, v_bf, nullptr, 2*JPAD, 512, 768);
  // attention (bf16 in, fp32 out)
  attention_wmma<<<dim3(4096/64, 8), dim3(128), 0, stream>>>(q_bf, k_bf, v_bf, attn);
  // out = attn @ w_out + b_out (fp32)    [4096,512] x [512,1024]
  gemm_af32<false><<<dim3(1024/64, 4096/64), blk, 0, stream>>>(attn, w_out, out, b_out, 4096, 1024, 512);
}
